// ObjectCentricCausal_25245817766359
// MI455X (gfx1250) — matrix operand order repaired and checked by reading, not compiled, not run on hardware
//
#include <hip/hip_runtime.h>
#include <hip/hip_bf16.h>

// ---------------------------------------------------------------------------
// ObjectCentricCausal forward for MI455X (gfx1250, wave32).
// GEMMs: v_wmma_f32_16x16x32_bf16, 2x2 tiles/wave, double-buffered LDS fed by
// GLOBAL_LOAD_ASYNC_TO_LDS_B128 (ASYNCcnt pipeline). Irregular stages on VALU.
// ---------------------------------------------------------------------------

#define N_OBJ    20
#define OBJ_DIM  64
#define D_MODEL  128
#define N_HEADS  4
#define HEAD_DIM 16
#define HID      (OBJ_DIM * N_OBJ)   // 1280
#define BATCH    4096

typedef __attribute__((ext_vector_type(16))) __bf16 bf16x16;
typedef __attribute__((ext_vector_type(8)))  float  f32x8;

__device__ __forceinline__ unsigned short f2bf(float f) {
    union { float f; unsigned int u; } a; a.f = f;
    unsigned int r = a.u + 0x7FFFu + ((a.u >> 16) & 1u);   // round-to-nearest-even
    return (unsigned short)(r >> 16);
}
__device__ __forceinline__ float bf2f(unsigned short h) {
    union { unsigned int u; float f; } a; a.u = ((unsigned int)h) << 16;
    return a.f;
}

// LDS byte offset of a __shared__ object: flat LDS addresses map via addr[31:0].
__device__ __forceinline__ unsigned lds_off(const void* p) {
    return (unsigned)(unsigned long long)p;
}

// Async global->LDS 16B copy (per lane), tracked by ASYNCcnt.
__device__ __forceinline__ void async_cp16(unsigned lds_byte_off, const unsigned short* gaddr) {
    asm volatile("global_load_async_to_lds_b128 %0, %1, off"
                 :: "v"(lds_byte_off), "v"(gaddr)
                 : "memory");
}

// ---------------------------------------------------------------------------
// GEMM: C[M,N] = opt_relu(A[M,K](bf16) @ Wb[N,ldw](bf16, cols wcol..)^T + bias)
// Block tile 128x64, BK=32, 256 threads = 8 waves, each wave a 32x32 output
// (2x2 WMMA tiles, 4 independent accumulators). A and W tiles double-buffered
// in LDS via async copies: issue tile k+1, s_wait_asynccnt 3 -> tile k ready.
// ---------------------------------------------------------------------------
#define LDSR 40                    // ushort row stride (80B, multiple of 16B)
#define A_TILE_US (128 * LDSR)
#define W_TILE_US (64 * LDSR)

template<bool HAS_CF, bool HAS_CB, bool HAS_BIAS, bool RELU>
__global__ __launch_bounds__(256)
void gemm_wmma_bf16(const unsigned short* __restrict__ A, int lda,
                    const unsigned short* __restrict__ Wb, int ldw, int wcol,
                    const float* __restrict__ bias,
                    float* __restrict__ Cf, unsigned short* __restrict__ Cb,
                    int M, int N, int K)
{
    __shared__ __align__(16) unsigned short As[2][A_TILE_US];  // 2x (128x32 bf16)
    __shared__ __align__(16) unsigned short Ws[2][W_TILE_US];  // 2x ( 64x32 bf16)

    const int tid  = threadIdx.x;
    const int lane = tid & 31;
    const int wave = tid >> 5;
    const int wm   = wave >> 1;           // 0..3 : 32-row band inside 128
    const int wn   = wave & 1;            // 0..1 : 32-col band inside 64
    const int mStart = blockIdx.y * 128;
    const int nStart = blockIdx.x * 64;

    const unsigned asBase = lds_off(&As[0][0]);
    const unsigned wsBase = lds_off(&Ws[0][0]);

    // per-thread tile-loader coordinates
    const int ar0 = (tid      ) >> 2, ac0 = ((tid      ) & 3) * 8;  // A chunk 0
    const int ar1 = (tid + 256) >> 2, ac1 = ((tid + 256) & 3) * 8;  // A chunk 1
    const int wr  = tid >> 2,         wc8 = (tid & 3) * 8;          // W chunk

    // issue one (A,W) tile pair: 3 async instructions per wave
    auto issue_tile = [&](int k0, int buf) {
        const unsigned ab = asBase + (unsigned)buf * (A_TILE_US * 2);
        const unsigned wb = wsBase + (unsigned)buf * (W_TILE_US * 2);
        async_cp16(ab + (unsigned)(ar0 * LDSR + ac0) * 2,
                   A + (size_t)(mStart + ar0) * lda + k0 + ac0);
        async_cp16(ab + (unsigned)(ar1 * LDSR + ac1) * 2,
                   A + (size_t)(mStart + ar1) * lda + k0 + ac1);
        async_cp16(wb + (unsigned)(wr * LDSR + wc8) * 2,
                   Wb + (size_t)(nStart + wr) * ldw + wcol + k0 + wc8);
    };

    f32x8 acc[2][2] = {};

    issue_tile(0, 0);
    int buf = 0;
    for (int k0 = 0; k0 < K; k0 += 32) {
        if (k0 + 32 < K) {
            issue_tile(k0 + 32, buf ^ 1);
            asm volatile("s_wait_asynccnt 0x3" ::: "memory");  // tile k landed
        } else {
            asm volatile("s_wait_asynccnt 0x0" ::: "memory");
        }
        __syncthreads();   // publish this wave-set's tile to all waves

        const unsigned short* Ab = &As[buf][0];
        const unsigned short* Wt = &Ws[buf][0];

        // ---- fragments per ISA 7.12.2 layouts ----
        union Frag { bf16x16 v; unsigned short u[16]; };
        Frag af[2], bfr[2];
        const int akb = (lane >> 4) * 8;           // 0 or 8
        #pragma unroll
        for (int t = 0; t < 2; ++t) {
            const int am = wm * 32 + t * 16 + (lane & 15);
            #pragma unroll
            for (int i = 0; i < 8; ++i) {
                af[t].u[i]     = Ab[am * LDSR + akb + i];        // K = akb..akb+7
                af[t].u[i + 8] = Ab[am * LDSR + 16 + akb + i];   // K = 16+akb..
            }
        }
        const int bkb = (lane >> 4) * 8;           // 0 or 8
        #pragma unroll
        for (int t = 0; t < 2; ++t) {
            const int bn = wn * 32 + t * 16 + (lane & 15);
            #pragma unroll
            for (int i = 0; i < 16; ++i)
                bfr[t].u[i] = Wt[bn * LDSR + (i < 8 ? bkb + i : 8 + bkb + i)];           // B[k][n] = W[n][k]
        }

        // 4 independent WMMAs per K-step
        #pragma unroll
        for (int i = 0; i < 2; ++i)
            #pragma unroll
            for (int j = 0; j < 2; ++j)
                acc[i][j] = __builtin_amdgcn_wmma_f32_16x16x32_bf16(
                    false, af[i].v, false, bfr[j].v, (short)0, acc[i][j], false, false);

        __syncthreads();   // all waves done reading buf before it is refilled
        buf ^= 1;
    }

    // ---- branch-free epilogue: VGPR r -> row r (lanes 0-15) / r+8 (16-31) ----
    #pragma unroll
    for (int j = 0; j < 2; ++j) {
        const int col = nStart + wn * 32 + j * 16 + (lane & 15);
        const float bv = HAS_BIAS ? bias[col] : 0.0f;
        #pragma unroll
        for (int i = 0; i < 2; ++i) {
            const int rbase = mStart + wm * 32 + i * 16 + ((lane < 16) ? 0 : 8);
            #pragma unroll
            for (int r = 0; r < 8; ++r) {
                float v = acc[i][j][r] + bv;
                if (RELU) v = fmaxf(v, 0.0f);
                size_t idx = (size_t)(rbase + r) * N + col;
                if (HAS_CF) Cf[idx] = v;
                if (HAS_CB) Cb[idx] = f2bf(v);
            }
        }
    }
}

// ---------------------------------------------------------------------------
// f32 -> bf16 elementwise (state + weight pre-pass)
// ---------------------------------------------------------------------------
__global__ __launch_bounds__(256)
void f32_to_bf16_kernel(const float* __restrict__ in, unsigned short* __restrict__ out, int n)
{
    int i = blockIdx.x * 256 + threadIdx.x;
    if (i < n) out[i] = f2bf(in[i]);
}

// ---------------------------------------------------------------------------
// Attention: one block per batch element, one wave per head, lanes 0..19 = rows.
// ---------------------------------------------------------------------------
__global__ __launch_bounds__(128)
void attention_kernel(const unsigned short* __restrict__ qkv,
                      unsigned short* __restrict__ ctx,
                      float* __restrict__ attn_out)
{
    __shared__ float qs[N_OBJ * 192];
    __shared__ float attn_s[N_HEADS * N_OBJ * N_OBJ];

    const int b   = blockIdx.x;
    const int tid = threadIdx.x;

    for (int i = tid; i < N_OBJ * 192; i += 128)
        qs[i] = bf2f(qkv[(size_t)b * N_OBJ * 192 + i]);
    __syncthreads();

    const int head = tid >> 5;
    const int lane = tid & 31;
    if (lane < N_OBJ) {
        const int n = lane;
        float qv[HEAD_DIM];
        #pragma unroll
        for (int d = 0; d < HEAD_DIM; ++d)
            qv[d] = qs[n * 192 + head * HEAD_DIM + d];

        float sc[N_OBJ];
        float mx = -1e30f;
        #pragma unroll
        for (int j = 0; j < N_OBJ; ++j) {
            float s = 0.f;
            #pragma unroll
            for (int d = 0; d < HEAD_DIM; ++d)
                s += qv[d] * qs[j * 192 + OBJ_DIM + head * HEAD_DIM + d];
            s *= 0.25f;                       // 1/sqrt(HEAD_DIM)
            sc[j] = s;
            mx = fmaxf(mx, s);
        }
        float sum = 0.f;
        #pragma unroll
        for (int j = 0; j < N_OBJ; ++j) { sc[j] = __expf(sc[j] - mx); sum += sc[j]; }
        const float inv = 1.0f / sum;
        #pragma unroll
        for (int j = 0; j < N_OBJ; ++j) {
            sc[j] *= inv;
            attn_s[head * (N_OBJ * N_OBJ) + n * N_OBJ + j] = sc[j];
        }
        #pragma unroll
        for (int d = 0; d < HEAD_DIM; ++d) {
            float c = 0.f;
            #pragma unroll
            for (int j = 0; j < N_OBJ; ++j)
                c += sc[j] * qs[j * 192 + 2 * OBJ_DIM + head * HEAD_DIM + d];
            ctx[(size_t)(b * N_OBJ + n) * OBJ_DIM + head * HEAD_DIM + d] = f2bf(c);
        }
    }
    __syncthreads();
    for (int i = tid; i < N_OBJ * N_OBJ; i += 128) {
        float m = 0.25f * (attn_s[i] + attn_s[400 + i] + attn_s[800 + i] + attn_s[1200 + i]);
        attn_out[(size_t)b * (N_OBJ * N_OBJ) + i] = m;
    }
}

// ---------------------------------------------------------------------------
// Per-object causal-property MLP: 64 -> relu(32) -> 3 (ph fused, no spills).
// ---------------------------------------------------------------------------
__global__ __launch_bounds__(128)
void cp_kernel(const float* __restrict__ objects,
               const float* __restrict__ w1, const float* __restrict__ b1,
               const float* __restrict__ w2, const float* __restrict__ b2,
               float* __restrict__ out)
{
    __shared__ float w1s[32 * 64];
    __shared__ float b1s[32];
    __shared__ float w2s[3 * 32];
    __shared__ float b2s[3];

    const int n   = blockIdx.y;
    const int tid = threadIdx.x;
    for (int i = tid; i < 32 * 64; i += 128) w1s[i] = w1[(size_t)n * 32 * 64 + i];
    if (tid < 32) b1s[tid] = b1[n * 32 + tid];
    if (tid < 96) w2s[tid] = w2[n * 96 + tid];
    if (tid < 3)  b2s[tid] = b2[n * 3 + tid];
    __syncthreads();

    const int b = blockIdx.x * 128 + tid;
    const float* obj = objects + (size_t)(b * N_OBJ + n) * OBJ_DIM;
    float o[OBJ_DIM];
    #pragma unroll
    for (int k = 0; k < OBJ_DIM; ++k) o[k] = obj[k];

    float a0 = b2s[0], a1 = b2s[1], a2 = b2s[2];
    for (int h = 0; h < 32; ++h) {
        float s = b1s[h];
        #pragma unroll
        for (int k = 0; k < OBJ_DIM; ++k) s += w1s[h * 64 + k] * o[k];
        s = fmaxf(s, 0.f);
        a0 += w2s[0 * 32 + h] * s;
        a1 += w2s[1 * 32 + h] * s;
        a2 += w2s[2 * 32 + h] * s;
    }
    float* dst = out + (size_t)(b * N_OBJ + n) * 3;
    dst[0] = a0; dst[1] = a1; dst[2] = a2;
}

// ---------------------------------------------------------------------------
// Pairwise interaction (drop diagonal); hi/hj tiles in LDS, block per batch.
// ---------------------------------------------------------------------------
__global__ __launch_bounds__(256)
void interact_kernel(const float* __restrict__ hi, const float* __restrict__ hj,
                     const float* __restrict__ b1, const float* __restrict__ w2,
                     const float* __restrict__ b2,
                     float* __restrict__ out)
{
    __shared__ float his[N_OBJ * OBJ_DIM];
    __shared__ float hjs[N_OBJ * OBJ_DIM];
    __shared__ float b1s[OBJ_DIM];
    __shared__ float w2s[OBJ_DIM];

    const int b   = blockIdx.x;
    const int tid = threadIdx.x;
    for (int i = tid; i < N_OBJ * OBJ_DIM; i += 256) {
        his[i] = hi[(size_t)b * N_OBJ * OBJ_DIM + i];
        hjs[i] = hj[(size_t)b * N_OBJ * OBJ_DIM + i];
    }
    if (tid < OBJ_DIM) { b1s[tid] = b1[tid]; w2s[tid] = w2[tid]; }
    __syncthreads();

    const float bias2 = b2[0];
    for (int p = tid; p < N_OBJ * N_OBJ; p += 256) {
        const int i = p / N_OBJ, j = p % N_OBJ;
        if (i == j) continue;
        float s = bias2;
        #pragma unroll
        for (int d = 0; d < OBJ_DIM; ++d)
            s += fmaxf(his[i * OBJ_DIM + d] + hjs[j * OBJ_DIM + d] + b1s[d], 0.f) * w2s[d];
        const int col = j - ((j > i) ? 1 : 0);
        out[(size_t)b * (N_OBJ * 19) + i * 19 + col] = s;
    }
}

// ---------------------------------------------------------------------------
// Host-side orchestration (all on `stream`, graph-capture safe).
// ---------------------------------------------------------------------------
extern "C" void kernel_launch(void* const* d_in, const int* in_sizes, int n_in,
                              void* d_out, int out_size, void* d_ws, size_t ws_size,
                              hipStream_t stream)
{
    (void)in_sizes; (void)n_in; (void)out_size; (void)ws_size;

    const float* state      = (const float*)d_in[0];
    const float* enc_w1     = (const float*)d_in[1];
    const float* enc_b1     = (const float*)d_in[2];
    const float* enc_w2     = (const float*)d_in[3];
    const float* enc_b2     = (const float*)d_in[4];
    const float* in_proj_w  = (const float*)d_in[5];
    const float* in_proj_b  = (const float*)d_in[6];
    const float* out_proj_w = (const float*)d_in[7];
    const float* out_proj_b = (const float*)d_in[8];
    const float* cp_w1      = (const float*)d_in[9];
    const float* cp_b1      = (const float*)d_in[10];
    const float* cp_w2      = (const float*)d_in[11];
    const float* cp_b2      = (const float*)d_in[12];
    const float* ip_w1      = (const float*)d_in[13];
    const float* ip_b1      = (const float*)d_in[14];
    const float* ip_w2      = (const float*)d_in[15];
    const float* ip_b2      = (const float*)d_in[16];

    // d_out regions (return order), all fp32
    float* out_objects = (float*)d_out;                        // B*20*64
    float* out_causal  = out_objects + (size_t)BATCH * N_OBJ * OBJ_DIM;
    float* out_props   = out_causal  + (size_t)BATCH * N_OBJ * OBJ_DIM;
    float* out_inter   = out_props   + (size_t)BATCH * N_OBJ * 3;
    float* out_attn    = out_inter   + (size_t)BATCH * N_OBJ * 19;

    // workspace layout (bf16 activations, bf16 weight copies, f32 hi/hj)
    unsigned short* ws_state = (unsigned short*)d_ws;                  // B*128
    unsigned short* ws_h     = ws_state + (size_t)BATCH * D_MODEL;     // B*1280
    unsigned short* ws_obj   = ws_h     + (size_t)BATCH * HID;         // B*1280
    unsigned short* ws_qkv   = ws_obj   + (size_t)BATCH * HID;         // B*20*192
    unsigned short* ws_ctx   = ws_qkv   + (size_t)BATCH * N_OBJ * 192; // B*20*64
    float*          ws_hi    = (float*)(ws_ctx + (size_t)BATCH * N_OBJ * OBJ_DIM);
    float*          ws_hj    = ws_hi + (size_t)BATCH * N_OBJ * OBJ_DIM;
    unsigned short* wb_enc1  = (unsigned short*)(ws_hj + (size_t)BATCH * N_OBJ * OBJ_DIM);
    unsigned short* wb_enc2  = wb_enc1 + (size_t)HID * D_MODEL;        // 1280*128
    unsigned short* wb_inp   = wb_enc2 + (size_t)HID * HID;            // 1280*1280
    unsigned short* wb_outp  = wb_inp  + (size_t)3 * OBJ_DIM * OBJ_DIM;// 192*64
    unsigned short* wb_ip    = wb_outp + (size_t)OBJ_DIM * OBJ_DIM;    // 64*64 -> 64*128

    const int M1 = BATCH;             // encoder rows
    const int M2 = BATCH * N_OBJ;     // object rows

    // 0) bf16 pre-pass: state + all GEMM weights
    f32_to_bf16_kernel<<<(BATCH * D_MODEL + 255) / 256, 256, 0, stream>>>(
        state, ws_state, BATCH * D_MODEL);
    f32_to_bf16_kernel<<<(HID * D_MODEL + 255) / 256, 256, 0, stream>>>(
        enc_w1, wb_enc1, HID * D_MODEL);
    f32_to_bf16_kernel<<<(HID * HID + 255) / 256, 256, 0, stream>>>(
        enc_w2, wb_enc2, HID * HID);
    f32_to_bf16_kernel<<<(3 * OBJ_DIM * OBJ_DIM + 255) / 256, 256, 0, stream>>>(
        in_proj_w, wb_inp, 3 * OBJ_DIM * OBJ_DIM);
    f32_to_bf16_kernel<<<(OBJ_DIM * OBJ_DIM + 255) / 256, 256, 0, stream>>>(
        out_proj_w, wb_outp, OBJ_DIM * OBJ_DIM);
    f32_to_bf16_kernel<<<(OBJ_DIM * 2 * OBJ_DIM + 255) / 256, 256, 0, stream>>>(
        ip_w1, wb_ip, OBJ_DIM * 2 * OBJ_DIM);

    // 1) h = relu(state @ enc_w1^T + b1)
    gemm_wmma_bf16<false, true, true, true>
        <<<dim3(HID / 64, M1 / 128), 256, 0, stream>>>(
        ws_state, D_MODEL, wb_enc1, D_MODEL, 0, enc_b1,
        nullptr, ws_h, M1, HID, D_MODEL);

    // 2) objects = h @ enc_w2^T + b2
    gemm_wmma_bf16<true, true, true, false>
        <<<dim3(HID / 64, M1 / 128), 256, 0, stream>>>(
        ws_h, HID, wb_enc2, HID, 0, enc_b2,
        out_objects, ws_obj, M1, HID, HID);

    // 3) qkv = objects @ in_proj_w^T + b
    gemm_wmma_bf16<false, true, true, false>
        <<<dim3(192 / 64, M2 / 128), 256, 0, stream>>>(
        ws_obj, OBJ_DIM, wb_inp, OBJ_DIM, 0, in_proj_b,
        nullptr, ws_qkv, M2, 192, OBJ_DIM);

    // 4) attention -> ctx (bf16), head-mean attention weights
    attention_kernel<<<BATCH, 128, 0, stream>>>(ws_qkv, ws_ctx, out_attn);

    // 5) causal_matrix = ctx @ out_proj_w^T + b
    gemm_wmma_bf16<true, false, true, false>
        <<<dim3(OBJ_DIM / 64, M2 / 128), 256, 0, stream>>>(
        ws_ctx, OBJ_DIM, wb_outp, OBJ_DIM, 0, out_proj_b,
        out_causal, nullptr, M2, OBJ_DIM, OBJ_DIM);

    // 6) hi = objects @ ip_w1[:, :64]^T ; hj = objects @ ip_w1[:, 64:]^T
    gemm_wmma_bf16<true, false, false, false>
        <<<dim3(OBJ_DIM / 64, M2 / 128), 256, 0, stream>>>(
        ws_obj, OBJ_DIM, wb_ip, 2 * OBJ_DIM, 0, nullptr,
        ws_hi, nullptr, M2, OBJ_DIM, OBJ_DIM);
    gemm_wmma_bf16<true, false, false, false>
        <<<dim3(OBJ_DIM / 64, M2 / 128), 256, 0, stream>>>(
        ws_obj, OBJ_DIM, wb_ip, 2 * OBJ_DIM, OBJ_DIM, nullptr,
        ws_hj, nullptr, M2, OBJ_DIM, OBJ_DIM);

    // 7) per-object property MLP
    cp_kernel<<<dim3(BATCH / 128, N_OBJ), 128, 0, stream>>>(
        out_objects, cp_w1, cp_b1, cp_w2, cp_b2, out_props);

    // 8) pairwise interaction
    interact_kernel<<<BATCH, 256, 0, stream>>>(
        ws_hi, ws_hj, ip_b1, ip_w2, ip_b2, out_inter);
}
